// MultiHeadAttention_24404004175897
// MI455X (gfx1250) — matrix operand order repaired and checked by reading, not compiled, not run on hardware
//
#include <hip/hip_runtime.h>

// ---------------------------------------------------------------------------
// Multi-head attention forward for MI455X (gfx1250, wave32, WMMA bf16).
//   q/k/v projections: f32 in -> bf16 out, f32 WMMA accum
//   logits -> raw scores into attn_w region + partial softmax stats
//   rowstats: combine partial (max,sumexp) per row (log-sum-exp merge)
//   av: normalize-on-load (exp(x-M)*inv), finalize attn_w in place, WMMA attn@V
//   out = ctx @ Wo + bo (f32)
// attn_w is streamed through HBM only 3x (write raw, read+write normalized)
// instead of 4x -- the dominant cost at 23.3 TB/s.
// ---------------------------------------------------------------------------

constexpr int Bsz = 4, Ssz = 2048, Dsz = 1024, Hn = 16, DHsz = 64;
constexpr float NEGF = -3.402823466e38f;   // float32 finfo.min

typedef __bf16 bf16;
typedef bf16  v16bf __attribute__((ext_vector_type(16)));
typedef float v8f   __attribute__((ext_vector_type(8)));

union Frag { v16bf v; unsigned u[8]; };
union Acc  { v8f   v; float    f[8]; };

__device__ __forceinline__ unsigned short f2bf_bits(float x) {
  union { bf16 h; unsigned short s; } c; c.h = (bf16)x; return c.s;
}
__device__ __forceinline__ unsigned pack2(float a, float b) {
  union { bf16 h[2]; unsigned u; } p; p.h[0] = (bf16)a; p.h[1] = (bf16)b; return p.u;
}
// A-fragment (16x32, 16-bit): reg r holds bf16 pair at k = base + 8*half
__device__ __forceinline__ int a_k(int r, int half) {
  return ((r & 4) ? (16 + ((r & 3) << 1)) : (r << 1)) + (half << 3);
}
// B-fragment (32x16, 16-bit): reg r holds bf16 pair at k = base + 8*half (same as A)
__device__ __forceinline__ int b_k(int r, int half) {
  return ((r & 4) ? (16 + ((r & 3) << 1)) : (r << 1)) + (half << 3);
}

// ---------------------------------------------------------------------------
// Weight transpose + f32 -> bf16 : WT[n][k] = bf16(W[k][n]),  D x D
// ---------------------------------------------------------------------------
__global__ __launch_bounds__(256)
void wtrans_kernel(const float* __restrict__ W, unsigned short* __restrict__ WT) {
  __shared__ float tile[32][33];
  const int bx = blockIdx.x * 32, by = blockIdx.y * 32;
  const int tx = threadIdx.x & 31, ty = threadIdx.x >> 5;   // 32 x 8 threads
  #pragma unroll
  for (int i = 0; i < 32; i += 8)
    tile[ty + i][tx] = W[(size_t)(by + ty + i) * Dsz + bx + tx];
  __syncthreads();
  #pragma unroll
  for (int i = 0; i < 32; i += 8)
    WT[(size_t)(bx + ty + i) * Dsz + by + tx] = f2bf_bits(tile[tx][ty + i]);
}

// ---------------------------------------------------------------------------
// GEMM + bias:  out[M x 1024] = A[M x 1024] @ WT^T + bias.
// WT is bf16, pre-transposed: WT[n][k]. A is f32 (converted to bf16 in LDS).
// Block: 256 thr = 8 waves, tile 128x128, K-steps of 32. Wave -> 32x64 (2x4 tiles).
// ---------------------------------------------------------------------------
__global__ __launch_bounds__(256)
void gemm_bias_kernel(const float* __restrict__ A,
                      const unsigned short* __restrict__ WT,
                      const float* __restrict__ bias,
                      unsigned short* __restrict__ outb,
                      float* __restrict__ outf, int M) {
  constexpr int K = Dsz, N = Dsz;
  __shared__ unsigned As[128][17];   // bf16 pairs, padded pitch (bank-conflict free)
  __shared__ unsigned Ws[128][17];
  const int t = threadIdx.x;
  const int m0 = blockIdx.x * 128, n0 = blockIdx.y * 128;
  const int w = t >> 5, lane = t & 31, half = lane >> 4, mn = lane & 15;
  const int rbase = (w >> 1) * 32, cbase = (w & 1) * 64;
  const unsigned* WT32 = (const unsigned*)WT;   // [N][K/2]

  Acc acc[2][4];
  #pragma unroll
  for (int i = 0; i < 2; ++i)
    #pragma unroll
    for (int j = 0; j < 4; ++j)
      #pragma unroll
      for (int e = 0; e < 8; ++e) acc[i][j].f[e] = 0.0f;

  for (int k0 = 0; k0 < K; k0 += 32) {
    #pragma unroll
    for (int it = 0; it < 8; ++it) {            // A tile: 128x32 f32 -> 2048 u32
      int idx = t + it * 256;
      int row = idx >> 4, kk = (idx & 15) << 1;
      float2 f2 = *(const float2*)(A + (size_t)(m0 + row) * K + k0 + kk);
      As[row][kk >> 1] = pack2(f2.x, f2.y);
    }
    #pragma unroll
    for (int it = 0; it < 8; ++it) {            // W tile: already bf16 pairs
      int idx = t + it * 256;
      int row = idx >> 4, kk2 = idx & 15;
      Ws[row][kk2] = WT32[(size_t)(n0 + row) * (K / 2) + (k0 >> 1) + kk2];
    }
    __syncthreads();
    Frag af[2], bfr[4];
    #pragma unroll
    for (int ri = 0; ri < 2; ++ri) {
      int row = rbase + ri * 16 + mn;
      #pragma unroll
      for (int r = 0; r < 8; ++r) af[ri].u[r] = As[row][a_k(r, half) >> 1];
    }
    #pragma unroll
    for (int ci = 0; ci < 4; ++ci) {
      int col = cbase + ci * 16 + mn;
      #pragma unroll
      for (int r = 0; r < 8; ++r) bfr[ci].u[r] = Ws[col][b_k(r, half) >> 1];
    }
    #pragma unroll
    for (int ri = 0; ri < 2; ++ri)
      #pragma unroll
      for (int ci = 0; ci < 4; ++ci)
        acc[ri][ci].v = __builtin_amdgcn_wmma_f32_16x16x32_bf16(
            false, af[ri].v, false, bfr[ci].v, (short)0, acc[ri][ci].v, false, false);
    __syncthreads();
  }

  // Uniform branch hoisted out of the store loops (no per-element cbranch).
  if (outb) {
    #pragma unroll
    for (int ri = 0; ri < 2; ++ri)
      #pragma unroll
      for (int ci = 0; ci < 4; ++ci) {
        int col = n0 + cbase + ci * 16 + mn;
        float bv = bias[col];
        #pragma unroll
        for (int r = 0; r < 8; ++r) {
          int row = m0 + rbase + ri * 16 + r + half * 8;
          outb[(size_t)row * N + col] = f2bf_bits(acc[ri][ci].f[r] + bv);
        }
      }
  } else {
    #pragma unroll
    for (int ri = 0; ri < 2; ++ri)
      #pragma unroll
      for (int ci = 0; ci < 4; ++ci) {
        int col = n0 + cbase + ci * 16 + mn;
        float bv = bias[col];
        #pragma unroll
        for (int r = 0; r < 8; ++r) {
          int row = m0 + rbase + ri * 16 + r + half * 8;
          outf[(size_t)row * N + col] = acc[ri][ci].f[r] + bv;
        }
      }
  }
  (void)M;
}

// ---------------------------------------------------------------------------
// logits[b,h,q,k] = (Q . K) / 8 + mask*NEGF   (raw scores, written to attn_w
// region) + partial softmax stats (max, sumexp) per (row, 64-col chunk).
// Stats layout: stats[((bh*S + q) * 32 + chunk) * 2] = {m, s}.
// ---------------------------------------------------------------------------
__global__ __launch_bounds__(256)
void logits_kernel(const unsigned short* __restrict__ Qb,
                   const unsigned short* __restrict__ Kb,
                   const float* __restrict__ mask, float* __restrict__ attn,
                   float* __restrict__ stats) {
  __shared__ unsigned Qs[128][33];   // 64 bf16 = 32 u32 (+1 pad) per row
  __shared__ unsigned Ks[128][33];
  const int t = threadIdx.x;
  const int q0 = blockIdx.x * 128, k0 = blockIdx.y * 128;
  const int bh = blockIdx.z, b = bh / Hn, h = bh % Hn;
  const unsigned* Qu = (const unsigned*)Qb;     // [B*S][D/2]
  const unsigned* Ku = (const unsigned*)Kb;
  const size_t rb = (size_t)b * Ssz;
  const int DW = Dsz / 2, hoff = h * (DHsz / 2);
  #pragma unroll
  for (int it = 0; it < 16; ++it) {
    int idx = t + it * 256;
    int row = idx >> 5, dd = idx & 31;
    Qs[row][dd] = Qu[(rb + q0 + row) * DW + hoff + dd];
    Ks[row][dd] = Ku[(rb + k0 + row) * DW + hoff + dd];
  }
  __syncthreads();
  const int w = t >> 5, lane = t & 31, half = lane >> 4, mn = lane & 15;
  const int rbase = (w >> 1) * 32, cbase = (w & 1) * 64;
  Acc acc[2][4];
  #pragma unroll
  for (int i = 0; i < 2; ++i)
    #pragma unroll
    for (int j = 0; j < 4; ++j)
      #pragma unroll
      for (int e = 0; e < 8; ++e) acc[i][j].f[e] = 0.0f;

  #pragma unroll
  for (int d0 = 0; d0 < 64; d0 += 32) {
    Frag af[2], bfr[4];
    #pragma unroll
    for (int ri = 0; ri < 2; ++ri) {
      int row = rbase + ri * 16 + mn;
      #pragma unroll
      for (int r = 0; r < 8; ++r) af[ri].u[r] = Qs[row][(a_k(r, half) + d0) >> 1];
    }
    #pragma unroll
    for (int ci = 0; ci < 4; ++ci) {
      int col = cbase + ci * 16 + mn;
      #pragma unroll
      for (int r = 0; r < 8; ++r) bfr[ci].u[r] = Ks[col][(b_k(r, half) + d0) >> 1];
    }
    #pragma unroll
    for (int ri = 0; ri < 2; ++ri)
      #pragma unroll
      for (int ci = 0; ci < 4; ++ci)
        acc[ri][ci].v = __builtin_amdgcn_wmma_f32_16x16x32_bf16(
            false, af[ri].v, false, bfr[ci].v, (short)0, acc[ri][ci].v, false, false);
  }

  const float scale = 0.125f;   // 1/sqrt(64)
  float* ap = attn + (size_t)bh * Ssz * Ssz;
  // Apply scale + mask in registers, store raw logits.
  #pragma unroll
  for (int ri = 0; ri < 2; ++ri)
    #pragma unroll
    for (int ci = 0; ci < 4; ++ci) {
      int kcol = k0 + cbase + ci * 16 + mn;
      #pragma unroll
      for (int r = 0; r < 8; ++r) {
        int qrow = q0 + rbase + ri * 16 + r + half * 8;
        float mv = mask[((size_t)b * Ssz + qrow) * Ssz + kcol];
        float v = acc[ri][ci].f[r] * scale + mv * NEGF;
        acc[ri][ci].f[r] = v;
        ap[(size_t)qrow * Ssz + kcol] = v;
      }
    }

  // Partial softmax stats. For a fixed (ri, r) each 16-lane half of the wave
  // holds 64 logits (4 ci x 16 mn) of one row -> reduce with xor 1/2/4/8
  // (stays inside the half on wave32).
  const int chunk = blockIdx.y * 2 + (w & 1);   // 64-col chunk id, 0..31
  #pragma unroll
  for (int ri = 0; ri < 2; ++ri) {
    #pragma unroll
    for (int r = 0; r < 8; ++r) {
      float m = NEGF;
      #pragma unroll
      for (int ci = 0; ci < 4; ++ci) m = fmaxf(m, acc[ri][ci].f[r]);
      float s = 0.0f;
      #pragma unroll
      for (int ci = 0; ci < 4; ++ci) s += __expf(acc[ri][ci].f[r] - m);
      #pragma unroll
      for (int off = 1; off <= 8; off <<= 1) {
        float mo = __shfl_xor(m, off, 32);
        float so = __shfl_xor(s, off, 32);
        float M = fmaxf(m, mo);
        s = s * __expf(m - M) + so * __expf(mo - M);
        m = M;
      }
      if (mn == 0) {
        int qrow = q0 + rbase + ri * 16 + r + half * 8;
        float2 st; st.x = m; st.y = s;
        *(float2*)(stats + (((size_t)bh * Ssz + qrow) * 32 + chunk) * 2) = st;
      }
    }
  }
}

// ---------------------------------------------------------------------------
// Combine 32 partial (m,s) per row -> rowM, rowInv = 1/sum. One wave32 per row.
// ---------------------------------------------------------------------------
__global__ __launch_bounds__(256)
void rowstats_kernel(const float* __restrict__ stats, float* __restrict__ rowM,
                     float* __restrict__ rowInv) {
  const int wid = threadIdx.x >> 5, lane = threadIdx.x & 31;
  const size_t row = (size_t)blockIdx.x * 8 + wid;
  float2 st = *(const float2*)(stats + (row * 32 + lane) * 2);
  float m = st.x, s = st.y;
  #pragma unroll
  for (int off = 16; off > 0; off >>= 1) {
    float mo = __shfl_xor(m, off, 32);
    float so = __shfl_xor(s, off, 32);
    float M = fmaxf(m, mo);
    s = s * __expf(m - M) + so * __expf(mo - M);
    m = M;
  }
  if (lane == 0) { rowM[row] = m; rowInv[row] = 1.0f / s; }
}

// ---------------------------------------------------------------------------
// ctx[b,q,h*64+n] = sum_k softmax(attn)[q,k] * V[b,k,h*64+n]
// Normalizes raw logits on load (exp(x-M)*inv), writes the probability back
// in place (finalizing attn_w), packs bf16 into LDS for the WMMAs.
// ---------------------------------------------------------------------------
__global__ __launch_bounds__(256)
void av_kernel(float* __restrict__ attn, const unsigned short* __restrict__ Vb,
               const float* __restrict__ rowM, const float* __restrict__ rowInv,
               float* __restrict__ ctx) {
  __shared__ unsigned As[128][17];
  __shared__ unsigned Vs[64][17];
  const int t = threadIdx.x;
  const int q0 = blockIdx.x * 128;
  const int bh = blockIdx.z, b = bh / Hn, h = bh % Hn;
  float* arow = attn + (size_t)bh * Ssz * Ssz;
  const unsigned short* vbase = Vb + (size_t)b * Ssz * Dsz + h * DHsz;
  const int w = t >> 5, lane = t & 31, half = lane >> 4, mn = lane & 15;
  const int rbase = (w >> 1) * 32, cbase = (w & 1) * 32;

  Acc acc[2][2];
  #pragma unroll
  for (int i = 0; i < 2; ++i)
    #pragma unroll
    for (int j = 0; j < 2; ++j)
      #pragma unroll
      for (int e = 0; e < 8; ++e) acc[i][j].f[e] = 0.0f;

  for (int k0 = 0; k0 < Ssz; k0 += 32) {
    #pragma unroll
    for (int it = 0; it < 8; ++it) {   // logits tile: normalize, write back, bf16
      int idx = t + it * 256;
      int row = idx >> 4, kk = (idx & 15) << 1;
      size_t grow = (size_t)bh * Ssz + q0 + row;
      float m = rowM[grow], inv = rowInv[grow];
      float* gp = arow + (size_t)(q0 + row) * Ssz + k0 + kk;
      float2 f2 = *(const float2*)gp;
      float p0 = __expf(f2.x - m) * inv;
      float p1 = __expf(f2.y - m) * inv;
      float2 pv; pv.x = p0; pv.y = p1;
      *(float2*)gp = pv;                       // finalize attn_w output
      As[row][kk >> 1] = pack2(p0, p1);
    }
    #pragma unroll
    for (int it = 0; it < 4; ++it) {            // V tile transposed: Vs[n][k]
      int idx = t + it * 256;
      int n = idx >> 4, kk = (idx & 15) << 1;
      unsigned lo = vbase[(size_t)(k0 + kk) * Dsz + n];
      unsigned hi = vbase[(size_t)(k0 + kk + 1) * Dsz + n];
      Vs[n][kk >> 1] = lo | (hi << 16);
    }
    __syncthreads();
    Frag af[2], bfr[2];
    #pragma unroll
    for (int ri = 0; ri < 2; ++ri) {
      int row = rbase + ri * 16 + mn;
      #pragma unroll
      for (int r = 0; r < 8; ++r) af[ri].u[r] = As[row][a_k(r, half) >> 1];
    }
    #pragma unroll
    for (int ci = 0; ci < 2; ++ci) {
      int col = cbase + ci * 16 + mn;
      #pragma unroll
      for (int r = 0; r < 8; ++r) bfr[ci].u[r] = Vs[col][b_k(r, half) >> 1];
    }
    #pragma unroll
    for (int ri = 0; ri < 2; ++ri)
      #pragma unroll
      for (int ci = 0; ci < 2; ++ci)
        acc[ri][ci].v = __builtin_amdgcn_wmma_f32_16x16x32_bf16(
            false, af[ri].v, false, bfr[ci].v, (short)0, acc[ri][ci].v, false, false);
    __syncthreads();
  }

  #pragma unroll
  for (int ri = 0; ri < 2; ++ri)
    #pragma unroll
    for (int ci = 0; ci < 2; ++ci) {
      int n = cbase + ci * 16 + mn;
      #pragma unroll
      for (int r = 0; r < 8; ++r) {
        int q = q0 + rbase + ri * 16 + r + half * 8;
        ctx[((size_t)b * Ssz + q) * Dsz + h * DHsz + n] = acc[ri][ci].f[r];
      }
    }
}

// ---------------------------------------------------------------------------
extern "C" void kernel_launch(void* const* d_in, const int* in_sizes, int n_in,
                              void* d_out, int out_size, void* d_ws, size_t ws_size,
                              hipStream_t stream) {
  const float* queries = (const float*)d_in[0];
  const float* keys    = (const float*)d_in[1];
  const float* values  = (const float*)d_in[2];
  const float* mask    = (const float*)d_in[3];
  const float* Wq = (const float*)d_in[4];  const float* bq = (const float*)d_in[5];
  const float* Wk = (const float*)d_in[6];  const float* bk = (const float*)d_in[7];
  const float* Wv = (const float*)d_in[8];  const float* bv = (const float*)d_in[9];
  const float* Wo = (const float*)d_in[10]; const float* bo = (const float*)d_in[11];

  float* out  = (float*)d_out;                       // (B,S,D)
  float* attn = out + (size_t)Bsz * Ssz * Dsz;       // (B,H,S,S)

  char* p = (char*)d_ws;
  unsigned short* Qbf = (unsigned short*)p; p += (size_t)Bsz * Ssz * Dsz * 2;
  unsigned short* Kbf = (unsigned short*)p; p += (size_t)Bsz * Ssz * Dsz * 2;
  unsigned short* Vbf = (unsigned short*)p; p += (size_t)Bsz * Ssz * Dsz * 2;
  float*          ctx = (float*)p;          p += (size_t)Bsz * Ssz * Dsz * 4;
  unsigned short* WqT = (unsigned short*)p; p += (size_t)Dsz * Dsz * 2;
  unsigned short* WkT = (unsigned short*)p; p += (size_t)Dsz * Dsz * 2;
  unsigned short* WvT = (unsigned short*)p; p += (size_t)Dsz * Dsz * 2;
  unsigned short* WoT = (unsigned short*)p; p += (size_t)Dsz * Dsz * 2;
  float* stats  = (float*)p;  p += (size_t)Bsz * Hn * Ssz * 32 * 2 * 4;
  float* rowM   = (float*)p;  p += (size_t)Bsz * Hn * Ssz * 4;
  float* rowInv = (float*)p;  p += (size_t)Bsz * Hn * Ssz * 4;

  dim3 tb(256);
  dim3 tg(Dsz / 32, Dsz / 32);
  wtrans_kernel<<<tg, tb, 0, stream>>>(Wq, WqT);
  wtrans_kernel<<<tg, tb, 0, stream>>>(Wk, WkT);
  wtrans_kernel<<<tg, tb, 0, stream>>>(Wv, WvT);
  wtrans_kernel<<<tg, tb, 0, stream>>>(Wo, WoT);

  const int M = Bsz * Ssz;                  // 8192
  dim3 gg(M / 128, Dsz / 128);
  gemm_bias_kernel<<<gg, tb, 0, stream>>>(queries, WqT, bq, Qbf, nullptr, M);
  gemm_bias_kernel<<<gg, tb, 0, stream>>>(keys,    WkT, bk, Kbf, nullptr, M);
  gemm_bias_kernel<<<gg, tb, 0, stream>>>(values,  WvT, bv, Vbf, nullptr, M);

  logits_kernel<<<dim3(Ssz / 128, Ssz / 128, Bsz * Hn), tb, 0, stream>>>(
      Qbf, Kbf, mask, attn, stats);
  rowstats_kernel<<<dim3(Bsz * Hn * Ssz / 8), tb, 0, stream>>>(stats, rowM, rowInv);
  av_kernel<<<dim3(Ssz / 128, 1, Bsz * Hn), tb, 0, stream>>>(attn, Vbf, rowM, rowInv, ctx);

  gemm_bias_kernel<<<gg, tb, 0, stream>>>(ctx, WoT, bo, nullptr, out, M);

  (void)in_sizes; (void)n_in; (void)out_size; (void)ws_size;
}
